// MultiHeadAttention_82265803588419
// MI455X (gfx1250) — compile-verified, matrix operand order checked by reading, not run on hardware
//
#include <hip/hip_runtime.h>
#include <hip/hip_bf16.h>

// ---------------------------------------------------------------------------
// MI455X / gfx1250 multi-head attention, all matmuls on v_wmma_f32_16x16x32_f16
// ---------------------------------------------------------------------------

typedef _Float16 f16;
typedef __attribute__((ext_vector_type(16))) _Float16 v16h;
typedef __attribute__((ext_vector_type(8)))  _Float16 v8h;
typedef __attribute__((ext_vector_type(8)))  float    v8f;

#define B_SZ     4
#define S_LEN    2048
#define D_MODEL  1024
#define NHEADS   16
#define HDIM     64
#define QKV_N    (3 * D_MODEL)   // 3072
#define NROWS    (B_SZ * S_LEN)  // 8192

// ---------------------------------------------------------------------------
// Fragment helpers (CDNA5 WMMA 16-bit A/B layout, wave32):
//   lane = hi*16 + r ; A(16x32): row = r, K chunks {hi*8..hi*8+7} and
//   {16+hi*8 .. 23+hi*8}  -> two contiguous 16-byte loads per lane.
//   B(32x16) mirrors A with N as the "row" index (we store B transposed).
//   C/D(16x16) f32: element e of v8f -> row = e + 8*hi, col = r.
// ---------------------------------------------------------------------------
__device__ __forceinline__ v16h ld_frag(const f16* __restrict__ p, int ld,
                                        int row0, int k0, int lane) {
    int r  = lane & 15;
    int hi = lane >> 4;
    const f16* q = p + (size_t)(row0 + r) * ld + k0 + hi * 8;
    union { v16h v; v8h h[2]; } u;
    u.h[0] = *(const v8h*)(q);
    u.h[1] = *(const v8h*)(q + 16);
    return u.v;
}

__device__ __forceinline__ v8f wmma16(v16h a, v16h b, v8f c) {
    return __builtin_amdgcn_wmma_f32_16x16x32_f16(
        /*neg_a=*/false, a, /*neg_b=*/false, b,
        /*c_mod=*/(short)0, c, /*reuse_a=*/false, /*reuse_b=*/false);
}

// ---------------------------------------------------------------------------
// Kernel 0a: f32 -> f16 elementwise
// ---------------------------------------------------------------------------
__global__ void k_cvt_f16(const float* __restrict__ src, f16* __restrict__ dst,
                          int n) {
    int i = blockIdx.x * blockDim.x + threadIdx.x;
    if (i < n) dst[i] = (f16)src[i];
}

// ---------------------------------------------------------------------------
// Kernel 0b: f32 [K][N] -> f16 transposed [N][K]
// ---------------------------------------------------------------------------
__global__ void k_transpose_f16(const float* __restrict__ src,
                                f16* __restrict__ dst, int K, int N) {
    int i = blockIdx.x * blockDim.x + threadIdx.x;
    if (i < K * N) {
        int k = i / N;
        int c = i - k * N;
        dst[(size_t)c * K + k] = (f16)src[i];
    }
}

// ---------------------------------------------------------------------------
// Kernel 1: QKV projection.  One wave per block; wave tile = 32 rows x 64 cols.
// C = Xh[8192,1024] * WqkvT'[col-major via transposed store] + b_qkv
// Epilogue scatters to Q[B,H,S,64], K[B,H,S,64], Vt[B,H,64,S]  (all f16).
// ---------------------------------------------------------------------------
__global__ void k_gemm_qkv(const f16* __restrict__ Xh,
                           const f16* __restrict__ WT,   // [3072][1024]
                           const float* __restrict__ bias,
                           f16* __restrict__ Qd, f16* __restrict__ Kd,
                           f16* __restrict__ Vt) {
    const int lane = threadIdx.x & 31;
    const int mt   = blockIdx.x % (NROWS / 32);
    const int nt   = blockIdx.x / (NROWS / 32);
    const int m0   = mt * 32;
    const int n0   = nt * 64;

    v8f acc[2][4];
#pragma unroll
    for (int mi = 0; mi < 2; ++mi)
#pragma unroll
        for (int ni = 0; ni < 4; ++ni) acc[mi][ni] = (v8f){};

    for (int k0 = 0; k0 < D_MODEL; k0 += 32) {
        __builtin_prefetch(Xh + (size_t)m0 * D_MODEL + k0 + 64, 0, 0);
        v16h a0 = ld_frag(Xh, D_MODEL, m0,      k0, lane);
        v16h a1 = ld_frag(Xh, D_MODEL, m0 + 16, k0, lane);
        v16h b[4];
#pragma unroll
        for (int ni = 0; ni < 4; ++ni)
            b[ni] = ld_frag(WT, D_MODEL, n0 + ni * 16, k0, lane);
#pragma unroll
        for (int ni = 0; ni < 4; ++ni) {
            acc[0][ni] = wmma16(a0, b[ni], acc[0][ni]);
            acc[1][ni] = wmma16(a1, b[ni], acc[1][ni]);
        }
    }

    const int r  = lane & 15;
    const int hi = lane >> 4;
#pragma unroll
    for (int mi = 0; mi < 2; ++mi)
#pragma unroll
        for (int ni = 0; ni < 4; ++ni) {
            const int col = n0 + ni * 16 + r;
            const int h   = col / (3 * HDIM);
            const int j   = col - h * (3 * HDIM);
            const float bv = bias[col];
#pragma unroll
            for (int e = 0; e < 8; ++e) {
                const int row = m0 + mi * 16 + e + 8 * hi;
                const int bb  = row >> 11;          // / S_LEN
                const int s   = row & (S_LEN - 1);
                const f16 hv  = (f16)(acc[mi][ni][e] + bv);
                if (j < HDIM) {
                    Qd[(((size_t)bb * NHEADS + h) * S_LEN + s) * HDIM + j] = hv;
                } else if (j < 2 * HDIM) {
                    Kd[(((size_t)bb * NHEADS + h) * S_LEN + s) * HDIM + (j - HDIM)] = hv;
                } else {
                    Vt[(((size_t)bb * NHEADS + h) * HDIM + (j - 2 * HDIM)) * S_LEN + s] = hv;
                }
            }
        }
}

// ---------------------------------------------------------------------------
// Kernel 2: flash attention.  128 threads = 4 waves; each wave owns one
// (b, h, 16-query) tile and streams 32-key chunks.  Probabilities round-trip
// through a per-wave LDS buffer to convert C-fragment layout -> A-fragment.
// ---------------------------------------------------------------------------
__global__ void k_attention(const f16* __restrict__ Qd,
                            const f16* __restrict__ Kd,
                            const f16* __restrict__ Vt,
                            const float* __restrict__ mask,
                            f16* __restrict__ vals) {
    __shared__ f16 sP[4][16 * 32];   // per-wave 16x32 probability staging

    const int wave = threadIdx.x >> 5;
    const int lane = threadIdx.x & 31;
    const int r    = lane & 15;
    const int hi   = lane >> 4;

    const int t  = blockIdx.x * 4 + wave;          // global 16-row q tile
    const int bb = t / (NHEADS * (S_LEN / 16));
    const int rm = t - bb * (NHEADS * (S_LEN / 16));
    const int h  = rm / (S_LEN / 16);
    const int q0 = (rm - h * (S_LEN / 16)) * 16;

    const f16* Qb  = Qd + ((size_t)bb * NHEADS + h) * S_LEN * HDIM;
    const f16* Kb  = Kd + ((size_t)bb * NHEADS + h) * S_LEN * HDIM;
    const f16* Vtb = Vt + ((size_t)bb * NHEADS + h) * HDIM * S_LEN;

    const v16h qa0 = ld_frag(Qb, HDIM, q0, 0,  lane);
    const v16h qa1 = ld_frag(Qb, HDIM, q0, 32, lane);

    float m_i[8], l_i[8];
#pragma unroll
    for (int e = 0; e < 8; ++e) { m_i[e] = -1e30f; l_i[e] = 0.0f; }
    v8f o[4];
#pragma unroll
    for (int ni = 0; ni < 4; ++ni) o[ni] = (v8f){};

    const float scale = 0.125f;   // 1/sqrt(64)

    for (int k0 = 0; k0 < S_LEN; k0 += 32) {
        // ---- scores: 16 x 32 tile = two 16x16 WMMA results
        v8f s0 = (v8f){}, s1 = (v8f){};
        {
            v16h kb00 = ld_frag(Kb, HDIM, k0,      0,  lane);
            v16h kb01 = ld_frag(Kb, HDIM, k0,      32, lane);
            v16h kb10 = ld_frag(Kb, HDIM, k0 + 16, 0,  lane);
            v16h kb11 = ld_frag(Kb, HDIM, k0 + 16, 32, lane);
            s0 = wmma16(qa0, kb00, s0);
            s0 = wmma16(qa1, kb01, s0);
            s1 = wmma16(qa0, kb10, s1);
            s1 = wmma16(qa1, kb11, s1);
        }

        // ---- scale + additive mask, per C-layout element
        float sc0[8], sc1[8];
#pragma unroll
        for (int e = 0; e < 8; ++e) {
            const size_t mrow = (size_t)(q0 + e + 8 * hi) * S_LEN;
            sc0[e] = s0[e] * scale + mask[mrow + k0 + r];
            sc1[e] = s1[e] * scale + mask[mrow + k0 + 16 + r];
        }

        // ---- row max over the 32-key chunk (reduce across 16-lane halves)
        float rmx[8];
#pragma unroll
        for (int e = 0; e < 8; ++e) rmx[e] = fmaxf(sc0[e], sc1[e]);
#pragma unroll
        for (int off = 1; off < 16; off <<= 1)
#pragma unroll
            for (int e = 0; e < 8; ++e)
                rmx[e] = fmaxf(rmx[e], __shfl_xor(rmx[e], off, 32));

        float alpha[8], p0[8], p1[8], rsum[8];
#pragma unroll
        for (int e = 0; e < 8; ++e) {
            const float mn = fmaxf(m_i[e], rmx[e]);
            alpha[e] = __expf(m_i[e] - mn);
            p0[e]    = __expf(sc0[e] - mn);
            p1[e]    = __expf(sc1[e] - mn);
            rsum[e]  = p0[e] + p1[e];
            m_i[e]   = mn;
        }
#pragma unroll
        for (int off = 1; off < 16; off <<= 1)
#pragma unroll
            for (int e = 0; e < 8; ++e)
                rsum[e] += __shfl_xor(rsum[e], off, 32);
#pragma unroll
        for (int e = 0; e < 8; ++e) l_i[e] = l_i[e] * alpha[e] + rsum[e];

        // ---- rescale running output
#pragma unroll
        for (int ni = 0; ni < 4; ++ni)
#pragma unroll
            for (int e = 0; e < 8; ++e) o[ni][e] *= alpha[e];

        // ---- C-layout -> A-layout via LDS
#pragma unroll
        for (int e = 0; e < 8; ++e) {
            const int row = e + 8 * hi;
            sP[wave][row * 32 + r]      = (f16)p0[e];
            sP[wave][row * 32 + 16 + r] = (f16)p1[e];
        }
        __syncthreads();
        const v16h pa = ld_frag(&sP[wave][0], 32, 0, 0, lane);

        // ---- O += P (16x32) * V (32x64); Vt rows are head dims
#pragma unroll
        for (int ni = 0; ni < 4; ++ni) {
            v16h vb = ld_frag(Vtb, S_LEN, ni * 16, k0, lane);
            o[ni] = wmma16(pa, vb, o[ni]);
        }
        __syncthreads();
    }

    // ---- normalize and emit vals[B,S,D] (col = h*64 + d) as f16
    float inv[8];
#pragma unroll
    for (int e = 0; e < 8; ++e) inv[e] = 1.0f / l_i[e];
#pragma unroll
    for (int ni = 0; ni < 4; ++ni)
#pragma unroll
        for (int e = 0; e < 8; ++e) {
            const int row = q0 + e + 8 * hi;
            const int col = h * HDIM + ni * 16 + r;
            vals[((size_t)bb * S_LEN + row) * D_MODEL + col] = (f16)(o[ni][e] * inv[e]);
        }
}

// ---------------------------------------------------------------------------
// Kernel 3: output projection.  out = vals[8192,1024] * W_out + b_out  (f32)
// ---------------------------------------------------------------------------
__global__ void k_gemm_out(const f16* __restrict__ Vh,
                           const f16* __restrict__ WT,   // [1024][1024]
                           const float* __restrict__ bias,
                           float* __restrict__ out) {
    const int lane = threadIdx.x & 31;
    const int mt   = blockIdx.x % (NROWS / 32);
    const int nt   = blockIdx.x / (NROWS / 32);
    const int m0   = mt * 32;
    const int n0   = nt * 64;

    v8f acc[2][4];
#pragma unroll
    for (int mi = 0; mi < 2; ++mi)
#pragma unroll
        for (int ni = 0; ni < 4; ++ni) acc[mi][ni] = (v8f){};

    for (int k0 = 0; k0 < D_MODEL; k0 += 32) {
        __builtin_prefetch(Vh + (size_t)m0 * D_MODEL + k0 + 64, 0, 0);
        v16h a0 = ld_frag(Vh, D_MODEL, m0,      k0, lane);
        v16h a1 = ld_frag(Vh, D_MODEL, m0 + 16, k0, lane);
        v16h b[4];
#pragma unroll
        for (int ni = 0; ni < 4; ++ni)
            b[ni] = ld_frag(WT, D_MODEL, n0 + ni * 16, k0, lane);
#pragma unroll
        for (int ni = 0; ni < 4; ++ni) {
            acc[0][ni] = wmma16(a0, b[ni], acc[0][ni]);
            acc[1][ni] = wmma16(a1, b[ni], acc[1][ni]);
        }
    }

    const int r  = lane & 15;
    const int hi = lane >> 4;
#pragma unroll
    for (int mi = 0; mi < 2; ++mi)
#pragma unroll
        for (int ni = 0; ni < 4; ++ni) {
            const int col  = n0 + ni * 16 + r;
            const float bv = bias[col];
#pragma unroll
            for (int e = 0; e < 8; ++e) {
                const int row = m0 + mi * 16 + e + 8 * hi;
                out[(size_t)row * D_MODEL + col] = acc[mi][ni][e] + bv;
            }
        }
}

// ---------------------------------------------------------------------------
// Host launcher
// ---------------------------------------------------------------------------
extern "C" void kernel_launch(void* const* d_in, const int* in_sizes, int n_in,
                              void* d_out, int out_size, void* d_ws, size_t ws_size,
                              hipStream_t stream) {
    const float* x      = (const float*)d_in[0];   // [4,2048,1024]
    const float* mask   = (const float*)d_in[1];   // [2048,2048]
    const float* W_qkv  = (const float*)d_in[2];   // [1024,3072]
    const float* b_qkv  = (const float*)d_in[3];   // [3072]
    const float* W_out  = (const float*)d_in[4];   // [1024,1024]
    const float* b_out  = (const float*)d_in[5];   // [1024]
    float* out          = (float*)d_out;           // [4,2048,1024]

    // workspace carve-up (f16 tensors)
    char* ws = (char*)d_ws;
    size_t off = 0;
    f16* xh    = (f16*)(ws + off); off += (size_t)NROWS * D_MODEL * 2;      // 16 MB
    f16* WqkvT = (f16*)(ws + off); off += (size_t)QKV_N * D_MODEL * 2;      //  6 MB
    f16* WoutT = (f16*)(ws + off); off += (size_t)D_MODEL * D_MODEL * 2;    //  2 MB
    f16* Qd    = (f16*)(ws + off); off += (size_t)NROWS * D_MODEL * 2;      // 16 MB
    f16* Kd    = (f16*)(ws + off); off += (size_t)NROWS * D_MODEL * 2;      // 16 MB
    f16* Vt    = (f16*)(ws + off); off += (size_t)NROWS * D_MODEL * 2;      // 16 MB
    f16* vals  = (f16*)(ws + off); off += (size_t)NROWS * D_MODEL * 2;      // 16 MB

    // 0) precision casts / transposes
    {
        int n = NROWS * D_MODEL;
        k_cvt_f16<<<(n + 255) / 256, 256, 0, stream>>>(x, xh, n);
        int nq = D_MODEL * QKV_N;
        k_transpose_f16<<<(nq + 255) / 256, 256, 0, stream>>>(W_qkv, WqkvT, D_MODEL, QKV_N);
        int no = D_MODEL * D_MODEL;
        k_transpose_f16<<<(no + 255) / 256, 256, 0, stream>>>(W_out, WoutT, D_MODEL, D_MODEL);
    }

    // 1) fused QKV projection -> Q, K, Vt (f16)
    {
        dim3 grid((NROWS / 32) * (QKV_N / 64));   // 256 * 48 = 12288 waves
        k_gemm_qkv<<<grid, 32, 0, stream>>>(xh, WqkvT, b_qkv, Qd, Kd, Vt);
    }

    // 2) flash attention -> vals (f16)
    {
        dim3 grid((B_SZ * NHEADS * (S_LEN / 16)) / 4);   // 8192 tiles / 4 waves
        k_attention<<<grid, 128, 0, stream>>>(Qd, Kd, Vt, mask, vals);
    }

    // 3) output projection -> out (f32)
    {
        dim3 grid((NROWS / 32) * (D_MODEL / 64));  // 256 * 16 = 4096 waves
        k_gemm_out<<<grid, 32, 0, stream>>>(vals, WoutT, b_out, out);
    }

    (void)in_sizes; (void)n_in; (void)out_size; (void)ws_size;
}
